// CustomLinearFullFP8_73538430042503
// MI455X (gfx1250) — 8-bit matrix operand order repaired and checked by reading, compiled, not run on hardware
//
#include <hip/hip_runtime.h>
#include <hip/hip_bf16.h>

typedef int   v16i __attribute__((ext_vector_type(16)));
typedef float v8f  __attribute__((ext_vector_type(8)));

#define FP8_MAX 448.0f

__device__ __forceinline__ unsigned char sw_f2e4m3(float f) {
    unsigned u = __float_as_uint(f);
    unsigned s = (u >> 31) << 7;
    float a = __uint_as_float(u & 0x7fffffffu);
    if (!(a >= 0.001953125f)) return (unsigned char)s;
    if (a >= 464.0f) return (unsigned char)(s | 0x7e);
    if (a < 0.015625f) {
        int k = (int)rintf(a * 512.0f);
        return (unsigned char)(s | (unsigned)k);
    }
    unsigned ab  = __float_as_uint(a);
    unsigned mant = ab & 0x7fffffu;
    unsigned exp  = ab >> 23;
    unsigned lsb  = (mant >> 20) & 1u;
    mant = mant + 0x7ffffu + lsb;
    if (mant >> 23) { exp++; mant = 0; }
    unsigned m3 = (mant >> 20) & 7u;
    int e8 = (int)exp - 127 + 7;
    if (e8 > 15) return (unsigned char)(s | 0x7e);
    return (unsigned char)(s | ((unsigned)e8 << 3) | m3);
}

__device__ __forceinline__ unsigned cvt4_fp8(float x0, float x1, float x2, float x3) {
#if __has_builtin(__builtin_amdgcn_cvt_pk_fp8_f32)
    int p = __builtin_amdgcn_cvt_pk_fp8_f32(x0, x1, 0, false);
    p     = __builtin_amdgcn_cvt_pk_fp8_f32(x2, x3, p, true);
    return (unsigned)p;
#else
    return (unsigned)sw_f2e4m3(x0) | ((unsigned)sw_f2e4m3(x1) << 8) |
           ((unsigned)sw_f2e4m3(x2) << 16) | ((unsigned)sw_f2e4m3(x3) << 24);
#endif
}

__global__ void k_init(unsigned* amax) {
    if (threadIdx.x < 2) amax[threadIdx.x] = 0u;
}

__global__ __launch_bounds__(256) void k_amax(const float4* __restrict__ p, long n4,
                                              unsigned* __restrict__ slot) {
    __shared__ float red[256];
    float m = 0.0f;
    long stride = (long)gridDim.x * blockDim.x;
    for (long i = (long)blockIdx.x * blockDim.x + threadIdx.x; i < n4; i += stride) {
        float4 v = p[i];
        m = fmaxf(m, fmaxf(fmaxf(fabsf(v.x), fabsf(v.y)), fmaxf(fabsf(v.z), fabsf(v.w))));
    }
    red[threadIdx.x] = m;
    __syncthreads();
    for (int s = 128; s > 0; s >>= 1) {
        if (threadIdx.x < s) red[threadIdx.x] = fmaxf(red[threadIdx.x], red[threadIdx.x + s]);
        __syncthreads();
    }
    if (threadIdx.x == 0) atomicMax(slot, __float_as_uint(red[0]));
}

__global__ __launch_bounds__(256) void k_quant_w(const float4* __restrict__ W,
                                                 unsigned* __restrict__ Wq,
                                                 const unsigned* __restrict__ amax, long n4) {
    float inv = FP8_MAX / __uint_as_float(amax[1]);
    long i = (long)blockIdx.x * blockDim.x + threadIdx.x;
    if (i < n4) {
        float4 v = W[i];
        Wq[i] = cvt4_fp8(v.x * inv, v.y * inv, v.z * inv, v.w * inv);
    }
}

__global__ __launch_bounds__(256) void k_gemm(const float* __restrict__ x,
                                              const unsigned char* __restrict__ Wq,
                                              float* __restrict__ out,
                                              const unsigned* __restrict__ amax) {
    __shared__ unsigned char sA[32 * 512];
    const int K = 512, N = 512;

    float ax = __uint_as_float(amax[0]);
    float aw = __uint_as_float(amax[1]);
    float inv_sx = FP8_MAX / ax;
    float scale  = (ax * aw) / (FP8_MAX * FP8_MAX);

    const int  tid   = threadIdx.x;
    const long mbase = (long)blockIdx.x * 32;

    const float4* xb = (const float4*)(x + mbase * K);
    unsigned* sA32 = (unsigned*)sA;
#pragma unroll
    for (int i = 0; i < 16; ++i) {
        int idx = tid + 256 * i;
        float4 v = xb[idx];
        sA32[idx] = cvt4_fp8(v.x * inv_sx, v.y * inv_sx, v.z * inv_sx, v.w * inv_sx);
    }
    __syncthreads();

    const int lane  = tid & 31;
    const int w     = tid >> 5;
    const int h     = lane >> 4;
    const int ln    = lane & 15;
    const int nbase = w * 64;

    v8f acc[2][4] = {};

#pragma unroll
    for (int ks = 0; ks < 4; ++ks) {
        v16i a[2];
#pragma unroll
        for (int r = 0; r < 2; ++r) {
            int row = r * 16 + ln;
#pragma unroll
            for (int half = 0; half < 2; ++half) {
#pragma unroll
                for (int p = 0; p < 4; ++p) {
                    int off = row * 512 + ks * 128 + half * 64 + (p >> 1) * 32 + h * 16 + (p & 1) * 8;
                    int2 t = *(const int2*)&sA[off];
                    a[r][half * 8 + p * 2]     = t.x;
                    a[r][half * 8 + p * 2 + 1] = t.y;
                }
            }
        }
#pragma unroll
        for (int t = 0; t < 4; ++t) {
            int c = nbase + t * 16 + ln;
            v16i b;
#pragma unroll
            for (int q = 0; q < 4; ++q) {
                const int4 v = *(const int4*)&Wq[(long)c * 512 + ks * 128 + q * 32 + h * 16];
                b[4 * q + 0] = v.x; b[4 * q + 1] = v.y;
                b[4 * q + 2] = v.z; b[4 * q + 3] = v.w;
            }
            acc[0][t] = __builtin_amdgcn_wmma_f32_16x16x128_fp8_fp8(a[0], b, (short)0, acc[0][t], false, false);
            acc[1][t] = __builtin_amdgcn_wmma_f32_16x16x128_fp8_fp8(a[1], b, (short)0, acc[1][t], false, false);
        }
    }

#pragma unroll
    for (int r = 0; r < 2; ++r) {
#pragma unroll
        for (int t = 0; t < 4; ++t) {
#pragma unroll
            for (int j = 0; j < 8; ++j) {
                long row = mbase + r * 16 + h * 8 + j;
                int  col = nbase + t * 16 + ln;
                __builtin_nontemporal_store(acc[r][t][j] * scale, &out[row * N + col]);
            }
        }
    }
}

extern "C" void kernel_launch(void* const* d_in, const int* in_sizes, int n_in,
                              void* d_out, int out_size, void* d_ws, size_t ws_size,
                              hipStream_t stream) {
    const float* x = (const float*)d_in[0];
    const float* W = (const float*)d_in[1];
    float* out = (float*)d_out;

    const int  K = 512, N = 512;
    const long M = (long)in_sizes[0] / K;

    unsigned*      amax = (unsigned*)d_ws;
    unsigned char* Wq   = (unsigned char*)d_ws + 256;

    const long xN4 = M * K / 4;
    const long wN4 = (long)N * K / 4;

    k_init<<<1, 32, 0, stream>>>(amax);
    k_amax<<<2048, 256, 0, stream>>>((const float4*)x, xN4, amax + 0);
    k_amax<<<64,   256, 0, stream>>>((const float4*)W, wN4, amax + 1);
    k_quant_w<<<(int)((wN4 + 255) / 256), 256, 0, stream>>>((const float4*)W, (unsigned*)Wq, amax, wN4);
    k_gemm<<<(int)(M / 32), 256, 0, stream>>>(x, Wq, out, amax);
}
